// Mamba_14705968021788
// MI455X (gfx1250) — compile-verified, matrix operand order checked by reading, not run on hardware
//
#include <hip/hip_runtime.h>
#include <hip/hip_bf16.h>
#include <math.h>

// ---------------- problem constants ----------------
#define D_MODEL  512
#define D_STATE  16
#define D_CONV   3
#define D_INNER  1024            // EXPAND * D_MODEL
#define DT_RANK  32              // ceil(512/16)
#define PROJ_DIM (DT_RANK + 4 * D_STATE + 1)   // 97
#define BATCH    2
#define SEQ      1024
#define M_ROWS   (BATCH * SEQ)   // 2048

typedef __attribute__((ext_vector_type(16))) _Float16 v16h;
typedef __attribute__((ext_vector_type(8)))  float    v8f;

union Frag16 {
    v16h v;
    uint4 q[2];
};

// ---------------- f32 -> f16 conversion ----------------
__global__ void k_f32_to_f16(const float* __restrict__ src,
                             _Float16* __restrict__ dst, int n) {
    int i = blockIdx.x * blockDim.x + threadIdx.x;
    if (i < n) dst[i] = (_Float16)src[i];
}

// ---------------- WMMA GEMM with async B staging --------------------------
// D(MxN) = A(MxK) * B^T, B given as Bt(NxK) row-major.
// Block = 256 threads = 8 waves, computes a 256(M) x 64(N) macro-tile:
//   - wave w owns rows [wave*32, wave*32+32): 2 M-tiles x 4 N-tiles = 8 accs
//   - the 64(N) x 32(K) B slab (4KB) is staged global->LDS with
//     global_load_async_to_lds_b128 (one b128 per thread), double-buffered,
//     synchronized with s_wait_asynccnt + barrier; waves read B fragments
//     back with ds_load_b128.
// Requires M % 256 == 0, N % 64 == 0, K % 32 == 0. EXEC is always full.
__global__ __launch_bounds__(256) void k_wmma_gemm_tiled(
    const _Float16* __restrict__ A,   // M x K row-major
    const _Float16* __restrict__ Bt,  // N x K row-major (B transposed)
    float* __restrict__ D,            // M x N row-major
    int M, int N, int K) {
    extern __shared__ char smem[];    // 2 x 4KB B-slab double buffer

    const int tid  = threadIdx.x;
    const int wave = tid >> 5;
    const int lane = tid & 31;
    const int half = lane & 15;
    const int hi   = lane >> 4;

    const int slabsN = N >> 6;
    const int bm = blockIdx.x / slabsN;
    const int bn = blockIdx.x % slabsN;
    const int mBase = bm * 256 + wave * 32;
    const int nBase = bn * 64;

    // staging role: thread t copies one 16B chunk of the B slab per K-step.
    // slab layout in LDS: 64 rows (N) x 64 bytes (32 f16 K-values), row-major
    const int srow = tid >> 2;                // 0..63
    const int schk = tid & 3;                 // 16B chunk within row
    const _Float16* sgp = Bt + (size_t)(nBase + srow) * K + schk * 8;
    const unsigned sldsoff = (unsigned)(srow * 64 + schk * 16);

    const _Float16* arow0 = A + (size_t)(mBase + half) * K;
    const _Float16* arow1 = A + (size_t)(mBase + 16 + half) * K;

    v8f acc[2][4] = {};
    const int kSteps = K >> 5;

    // stage slab 0
    {
        const _Float16* g = sgp;
        unsigned l = sldsoff;
        asm volatile("global_load_async_to_lds_b128 %0, %1, off"
                     :: "v"(l), "v"(g) : "memory");
    }
    asm volatile("s_wait_asynccnt 0x0" ::: "memory");
    __syncthreads();

    for (int ks = 0; ks < kSteps; ++ks) {
        const int k0  = ks << 5;
        const int cur = ks & 1;

        // stage next slab into the other buffer (async, overlapped with math)
        if (ks + 1 < kSteps) {
            const _Float16* g = sgp + k0 + 32;
            unsigned l = sldsoff + (unsigned)((cur ^ 1) * 4096);
            asm volatile("global_load_async_to_lds_b128 %0, %1, off"
                         :: "v"(l), "v"(g) : "memory");
            __builtin_prefetch(arow0 + k0 + 32, 0, 1);   // A next slab hint
            __builtin_prefetch(arow1 + k0 + 32, 0, 1);
        }

        // A fragments straight from global (per-wave, L2-resident)
        Frag16 a0, a1;
        a0.q[0] = *(const uint4*)(arow0 + k0 + hi * 8);
        a0.q[1] = *(const uint4*)(arow0 + k0 + 16 + hi * 8);
        a1.q[0] = *(const uint4*)(arow1 + k0 + hi * 8);
        a1.q[1] = *(const uint4*)(arow1 + k0 + 16 + hi * 8);

        // B fragments from LDS slab; 8 WMMAs per K-step
        const char* sbase = smem + cur * 4096;
#pragma unroll
        for (int j = 0; j < 4; ++j) {
            Frag16 b;
            b.q[0] = *(const uint4*)(sbase + (j * 16 + half) * 64 + hi * 16);
            b.q[1] = *(const uint4*)(sbase + (j * 16 + half) * 64 + 32 + hi * 16);
            acc[0][j] = __builtin_amdgcn_wmma_f32_16x16x32_f16(
                false, a0.v, false, b.v, (short)0, acc[0][j], false, false);
            acc[1][j] = __builtin_amdgcn_wmma_f32_16x16x32_f16(
                false, a1.v, false, b.v, (short)0, acc[1][j], false, false);
        }

        // next-slab staging complete + all waves done reading cur
        asm volatile("s_wait_asynccnt 0x0" ::: "memory");
        __syncthreads();
    }

    // D layout: VGPR r -> row m0 + hi*8 + r, col n0 + half
#pragma unroll
    for (int mi = 0; mi < 2; ++mi) {
#pragma unroll
        for (int j = 0; j < 4; ++j) {
            float* drow = D + (size_t)(mBase + mi * 16 + hi * 8) * N
                            + nBase + j * 16 + half;
#pragma unroll
            for (int r = 0; r < 8; ++r) drow[(size_t)r * N] = acc[mi][j][r];
        }
    }
}

// ---------------- causal depthwise conv1d + SiLU ----------------
__global__ void k_conv_silu(const float* __restrict__ xz,
                            const float* __restrict__ conv_w,
                            const float* __restrict__ conv_b,
                            float* __restrict__ xc) {
    int idx = blockIdx.x * blockDim.x + threadIdx.x;
    if (idx >= M_ROWS * D_INNER) return;
    int d = idx % D_INNER;
    int m = idx / D_INNER;
    int l = m % SEQ;
    int b = m / SEQ;

    float acc = conv_b[d];
#pragma unroll
    for (int j = 0; j < D_CONV; ++j) {
        int lp = l - (D_CONV - 1) + j;
        if (lp >= 0) {
            acc += conv_w[d * D_CONV + j] *
                   xz[((size_t)(b * SEQ + lp)) * (2 * D_INNER) + d];
        }
    }
    float s = acc / (1.0f + __expf(-acc));   // SiLU
    xc[idx] = s;
}

// ---------------- x_proj: proj(M x 97) = xc(M x 1024) * x_proj_w^T ----------
__global__ void k_xproj(const float* __restrict__ xc,
                        const float* __restrict__ xpw,   // (97, 1024)
                        float* __restrict__ proj) {
    int idx = blockIdx.x * blockDim.x + threadIdx.x;
    if (idx >= M_ROWS * PROJ_DIM) return;
    int c = idx % PROJ_DIM;
    int m = idx / PROJ_DIM;
    const float* xr = xc + (size_t)m * D_INNER;
    const float* wr = xpw + (size_t)c * D_INNER;
    float acc = 0.f;
#pragma unroll 4
    for (int k = 0; k < D_INNER; ++k) acc = fmaf(xr[k], wr[k], acc);
    proj[idx] = acc;
}

// ---------------- delta = softplus(delta_raw @ dt_w^T + dt_b) ----------------
__global__ void k_delta(const float* __restrict__ proj,
                        const float* __restrict__ dt_w,   // (1024, 32)
                        const float* __restrict__ dt_b,   // (1024,)
                        float* __restrict__ delta) {
    int idx = blockIdx.x * blockDim.x + threadIdx.x;
    if (idx >= M_ROWS * D_INNER) return;
    int d = idx % D_INNER;
    int m = idx / D_INNER;
    const float* pr = proj + (size_t)m * PROJ_DIM;
    const float* wr = dt_w + (size_t)d * DT_RANK;
    float acc = dt_b[d];
#pragma unroll
    for (int r = 0; r < DT_RANK; ++r) acc = fmaf(pr[r], wr[r], acc);
    delta[idx] = (acc > 20.f) ? acc : log1pf(__expf(acc));   // softplus
}

// ---------------- complex selective scan (trapezoidal) ----------------
// One (b,d) pair per 16-lane half-wave; lane within half = state index n.
__global__ __launch_bounds__(256) void k_scan(
    const float* __restrict__ delta,   // (M, D_INNER)
    const float* __restrict__ xc,      // (M, D_INNER)
    const float* __restrict__ proj,    // (M, 97)
    const float* __restrict__ A_log,   // (D_INNER, 16)
    const float* __restrict__ A_imag,  // (D_INNER, 16)
    const float* __restrict__ D_skip,  // (D_INNER,)
    float* __restrict__ y) {           // (M, D_INNER)
    const int tid   = threadIdx.x;
    const int wave  = tid >> 5;
    const int lane  = tid & 31;
    const int group = lane >> 4;
    const int n     = lane & 15;

    const int pair = (blockIdx.x * 8 + wave) * 2 + group;
    if (pair >= BATCH * D_INNER) return;
    const int b = pair / D_INNER;
    const int d = pair % D_INNER;

    const float Ar  = -__expf(A_log[d * D_STATE + n]);
    const float Ai  = A_imag[d * D_STATE + n];
    const float Dsk = D_skip[d];

    float hr = 0.f, hi_ = 0.f;          // complex state h
    float bxpr = 0.f, bxpi = 0.f;       // Bx_{t-1}

    for (int t = 0; t < SEQ; ++t) {
        const size_t row = (size_t)(b * SEQ + t);
        const float dt_v = delta[row * D_INNER + d];
        const float xc_v = xc[row * D_INNER + d];
        const float* pr  = proj + row * PROJ_DIM;
        const float Br   = pr[DT_RANK + n];
        const float Bi   = pr[DT_RANK + D_STATE + n];
        const float Cr   = pr[DT_RANK + 2 * D_STATE + n];
        const float Ci   = pr[DT_RANK + 3 * D_STATE + n];
        const float lamr = pr[DT_RANK + 4 * D_STATE];
        const float lam  = 1.0f / (1.0f + __expf(-lamr));

        float si, co;
        __sincosf(dt_v * Ai, &si, &co);
        const float er = __expf(dt_v * Ar);
        const float ar = er * co;
        const float ai = er * si;

        const float bxr = Br * xc_v;
        const float bxi = Bi * xc_v;

        const float s  = (1.0f - lam) * dt_v;
        const float br = s * ar, bi = s * ai;
        const float g  = lam * dt_v;

        const float ur = (br * bxpr - bi * bxpi) + g * bxr;
        const float ui = (br * bxpi + bi * bxpr) + g * bxi;

        const float nhr = ar * hr - ai * hi_ + ur;
        const float nhi = ar * hi_ + ai * hr + ui;
        hr = nhr; hi_ = nhi;
        bxpr = bxr; bxpi = bxi;

        float v = hr * Cr + hi_ * Ci;
        v += __shfl_xor(v, 8, 16);
        v += __shfl_xor(v, 4, 16);
        v += __shfl_xor(v, 2, 16);
        v += __shfl_xor(v, 1, 16);
        if (n == 0) y[row * D_INNER + d] = v + Dsk * xc_v;
    }
}

// ---------------- gate: g = y * silu(z), convert to f16 ----------------
__global__ void k_gate_to_f16(const float* __restrict__ y,
                              const float* __restrict__ xz,
                              _Float16* __restrict__ g) {
    int idx = blockIdx.x * blockDim.x + threadIdx.x;
    if (idx >= M_ROWS * D_INNER) return;
    int d = idx % D_INNER;
    int m = idx / D_INNER;
    float z = xz[(size_t)m * (2 * D_INNER) + D_INNER + d];
    float sz = z / (1.0f + __expf(-z));
    g[idx] = (_Float16)(y[idx] * sz);
}

// ---------------- launch ----------------
extern "C" void kernel_launch(void* const* d_in, const int* in_sizes, int n_in,
                              void* d_out, int out_size, void* d_ws, size_t ws_size,
                              hipStream_t stream) {
    const float* x         = (const float*)d_in[0];   // (2,1024,512)
    const float* in_proj_w = (const float*)d_in[1];   // (2048,512)
    const float* conv_w    = (const float*)d_in[2];   // (1024,1,3)
    const float* conv_b    = (const float*)d_in[3];   // (1024,)
    const float* x_proj_w  = (const float*)d_in[4];   // (97,1024)
    const float* dt_w      = (const float*)d_in[5];   // (1024,32)
    const float* dt_b      = (const float*)d_in[6];   // (1024,)
    const float* A_log     = (const float*)d_in[7];   // (1024,16)
    const float* A_imag    = (const float*)d_in[8];   // (1024,16)
    const float* D_skip    = (const float*)d_in[9];   // (1024,)
    const float* out_proj_w= (const float*)d_in[10];  // (512,1024)
    float* out = (float*)d_out;                       // (2,1024,512)

    char* p = (char*)d_ws;
    auto bump = [&p](size_t bytes) -> char* {
        char* r = p;
        p += (bytes + 255) & ~(size_t)255;
        return r;
    };
    _Float16* Xh   = (_Float16*)bump((size_t)M_ROWS * D_MODEL * 2);
    _Float16* Wih  = (_Float16*)bump((size_t)(2 * D_INNER) * D_MODEL * 2);
    _Float16* Woh  = (_Float16*)bump((size_t)D_MODEL * D_INNER * 2);
    _Float16* Gh   = (_Float16*)bump((size_t)M_ROWS * D_INNER * 2);
    float*    xz   = (float*)bump((size_t)M_ROWS * 2 * D_INNER * 4);
    float*    xcb  = (float*)bump((size_t)M_ROWS * D_INNER * 4);
    float*    proj = (float*)bump((size_t)M_ROWS * PROJ_DIM * 4);
    float*    dlt  = (float*)bump((size_t)M_ROWS * D_INNER * 4);
    float*    yb   = (float*)bump((size_t)M_ROWS * D_INNER * 4);

    const int T = 256;
    const size_t LDS_BYTES = 8192;   // 2 x 4KB B-slab double buffer

    // 1) convert activations + weights to f16
    {
        int n = M_ROWS * D_MODEL;
        k_f32_to_f16<<<(n + T - 1) / T, T, 0, stream>>>(x, Xh, n);
        n = 2 * D_INNER * D_MODEL;
        k_f32_to_f16<<<(n + T - 1) / T, T, 0, stream>>>(in_proj_w, Wih, n);
        n = D_MODEL * D_INNER;
        k_f32_to_f16<<<(n + T - 1) / T, T, 0, stream>>>(out_proj_w, Woh, n);
    }

    // 2) xz = x @ in_proj_w^T : M=2048, N=2048, K=512
    {
        int blocks = (M_ROWS / 256) * ((2 * D_INNER) / 64);   // 8*32 = 256
        k_wmma_gemm_tiled<<<blocks, 256, LDS_BYTES, stream>>>(
            Xh, Wih, xz, M_ROWS, 2 * D_INNER, D_MODEL);
    }

    // 3) causal depthwise conv + SiLU
    {
        int n = M_ROWS * D_INNER;
        k_conv_silu<<<(n + T - 1) / T, T, 0, stream>>>(xz, conv_w, conv_b, xcb);
    }

    // 4) proj = xc @ x_proj_w^T  (M x 97)
    {
        int n = M_ROWS * PROJ_DIM;
        k_xproj<<<(n + T - 1) / T, T, 0, stream>>>(xcb, x_proj_w, proj);
    }

    // 5) delta = softplus(delta_raw @ dt_w^T + dt_b)
    {
        int n = M_ROWS * D_INNER;
        k_delta<<<(n + T - 1) / T, T, 0, stream>>>(proj, dt_w, dt_b, dlt);
    }

    // 6) complex selective scan -> y (includes D-skip)
    {
        int pairs = BATCH * D_INNER;            // 2048, 16 per block
        int blocks = (pairs + 15) / 16;         // 128
        k_scan<<<blocks, 256, 0, stream>>>(dlt, xcb, proj,
                                           A_log, A_imag, D_skip, yb);
    }

    // 7) gate with silu(z), convert to f16
    {
        int n = M_ROWS * D_INNER;
        k_gate_to_f16<<<(n + T - 1) / T, T, 0, stream>>>(yb, xz, Gh);
    }

    // 8) out = g @ out_proj_w^T : M=2048, N=512, K=1024
    {
        int blocks = (M_ROWS / 256) * (D_MODEL / 64);   // 8*8 = 64
        k_wmma_gemm_tiled<<<blocks, 256, LDS_BYTES, stream>>>(
            Gh, Woh, out, M_ROWS, D_MODEL, D_INNER);
    }
}
